// AINN_SC_Merged_26774826123369
// MI455X (gfx1250) — matrix operand order repaired and checked by reading, not compiled, not run on hardware
//
#include <hip/hip_runtime.h>
#include <hip/hip_bf16.h>
#include <stdint.h>

// ---- sizes ----
#define BSZ   8
#define TT    61
#define FEAT  13
#define DD    32
#define HID   96
#define RR    256
#define TP    64      // T padded to 64 for WMMA

typedef __attribute__((ext_vector_type(16))) _Float16 v16h;
typedef __attribute__((ext_vector_type(8)))  float    v8f;

__device__ __forceinline__ float gelu_f(float x) {
    return 0.5f * x * (1.0f + erff(x * 0.70710678118654752440f));
}

// ---------------------------------------------------------------------------
// Kernel 1: encoder for x (8 seqs) and refs (256 seqs). One block per sequence.
// Produces f16 qW (B,64,32) and f16 kref (R,64,32), rows 61..63 zero-padded.
// ---------------------------------------------------------------------------
__global__ __launch_bounds__(128) void ainn_enc_kernel(
    const float* __restrict__ x, const float* __restrict__ refs,
    const float* __restrict__ prj_w, const float* __restrict__ prj_b,
    const float* __restrict__ b1_dw_w, const float* __restrict__ b1_dw_b,
    const float* __restrict__ b1_pw_w, const float* __restrict__ b1_pw_b,
    const float* __restrict__ b2_dw_w, const float* __restrict__ b2_dw_b,
    const float* __restrict__ b2_pw_w, const float* __restrict__ b2_pw_b,
    const float* __restrict__ b3_dw_w, const float* __restrict__ b3_dw_b,
    const float* __restrict__ b3_pw_w, const float* __restrict__ b3_pw_b,
    const float* __restrict__ ln_g, const float* __restrict__ ln_b,
    const float* __restrict__ nn1_W,
    _Float16* __restrict__ qWh, _Float16* __restrict__ krefh)
{
    __shared__ float xs[TT * FEAT];
    __shared__ float hb[TT * HID];
    __shared__ float yb[TT * HID];
    __shared__ float ob[TT * DD];

    const int s   = blockIdx.x;
    const int tid = threadIdx.x;
    const float* xin = (s < BSZ) ? (x + (size_t)s * TT * FEAT)
                                 : (refs + (size_t)(s - BSZ) * TT * FEAT);

    for (int i = tid; i < TT * FEAT; i += 128) xs[i] = xin[i];
    __syncthreads();

    // projection 13 -> 96
    for (int i = tid; i < TT * HID; i += 128) {
        int t = i / HID, c = i % HID;
        float a = prj_b[c];
        #pragma unroll
        for (int f = 0; f < FEAT; ++f) a += xs[t * FEAT + f] * prj_w[c * FEAT + f];
        hb[i] = a;
    }
    __syncthreads();

    // residual conv blocks 1 and 2
    for (int blk = 0; blk < 2; ++blk) {
        const float* dww = blk ? b2_dw_w : b1_dw_w;
        const float* dwb = blk ? b2_dw_b : b1_dw_b;
        const float* pww = blk ? b2_pw_w : b1_pw_w;
        const float* pwb = blk ? b2_pw_b : b1_pw_b;
        const int dil = blk ? 2 : 1;
        for (int i = tid; i < TT * HID; i += 128) {
            int t = i / HID, c = i % HID;
            float a = dwb[c];
            #pragma unroll
            for (int j = 0; j < 7; ++j) {
                int tt2 = t + (j - 3) * dil;
                if (tt2 >= 0 && tt2 < TT) a += hb[tt2 * HID + c] * dww[c * 7 + j];
            }
            yb[i] = a;
        }
        __syncthreads();
        for (int i = tid; i < TT * HID; i += 128) {
            int t = i / HID, c = i % HID;
            float a = pwb[c];
            const float* wr = pww + c * HID;
            for (int k = 0; k < HID; ++k) a += yb[t * HID + k] * wr[k];
            hb[i] = gelu_f(a) + hb[i];
        }
        __syncthreads();
    }

    // block 3: dw dil=4, pw 96 -> 32, gelu (no residual)
    for (int i = tid; i < TT * HID; i += 128) {
        int t = i / HID, c = i % HID;
        float a = b3_dw_b[c];
        #pragma unroll
        for (int j = 0; j < 7; ++j) {
            int tt2 = t + (j - 3) * 4;
            if (tt2 >= 0 && tt2 < TT) a += hb[tt2 * HID + c] * b3_dw_w[c * 7 + j];
        }
        yb[i] = a;
    }
    __syncthreads();
    for (int i = tid; i < TT * DD; i += 128) {
        int t = i / DD, c = i % DD;
        float a = b3_pw_b[c];
        const float* wr = b3_pw_w + c * HID;
        for (int k = 0; k < HID; ++k) a += yb[t * HID + k] * wr[k];
        ob[i] = gelu_f(a);
    }
    __syncthreads();

    // layernorm over D then L2-normalize, per time step
    if (tid < TT) {
        const int t = tid;
        float m = 0.f;
        #pragma unroll
        for (int d = 0; d < DD; ++d) m += ob[t * DD + d];
        m *= (1.f / DD);
        float v = 0.f;
        #pragma unroll
        for (int d = 0; d < DD; ++d) { float dd = ob[t * DD + d] - m; v += dd * dd; }
        v *= (1.f / DD);
        const float iv = rsqrtf(v + 1e-5f);
        float tmp[DD];
        float nrm = 0.f;
        #pragma unroll
        for (int d = 0; d < DD; ++d) {
            float z = (ob[t * DD + d] - m) * iv * ln_g[d] + ln_b[d];
            tmp[d] = z; nrm += z * z;
        }
        const float il = 1.f / fmaxf(sqrtf(nrm), 1e-12f);
        #pragma unroll
        for (int d = 0; d < DD; ++d) ob[t * DD + d] = tmp[d] * il;
    }
    __syncthreads();

    if (s < BSZ) {
        _Float16* dst = qWh + (size_t)s * TP * DD;
        for (int i = tid; i < TT * DD; i += 128) {
            int t = i / DD, f = i % DD;
            float a = 0.f;
            #pragma unroll
            for (int d = 0; d < DD; ++d)
                a += ob[t * DD + d] * 0.5f * (nn1_W[d * DD + f] + nn1_W[f * DD + d]);
            dst[t * DD + f] = (_Float16)a;
        }
        for (int i = tid + TT * DD; i < TP * DD; i += 128) dst[i] = (_Float16)0.f;
    } else {
        _Float16* dst = krefh + (size_t)(s - BSZ) * TP * DD;
        for (int i = tid; i < TT * DD; i += 128) dst[i] = (_Float16)ob[i];
        for (int i = tid + TT * DD; i < TP * DD; i += 128) dst[i] = (_Float16)0.f;
    }
}

// ---------------------------------------------------------------------------
// Kernel 2: fused C-tile (WMMA f16->f32) + wavefront DP. One 64-thread block
// (2 wave32) per (b,r) pair. C tile: 4x4 grid of 16x16 WMMAs, K=32 in one shot.
// ---------------------------------------------------------------------------
__global__ __launch_bounds__(64) void ainn_dp_kernel(
    const _Float16* __restrict__ qWh, const _Float16* __restrict__ krefh,
    const float* __restrict__ nn1_scale,
    const float* __restrict__ nn2_ln_g, const float* __restrict__ nn2_ln_b,
    const float* __restrict__ nn2_w1, const float* __restrict__ nn2_b1,
    const float* __restrict__ nn2_w2, const float* __restrict__ nn2_b2,
    float* __restrict__ flat)
{
    __shared__ _Float16 lA[TP * DD];
    __shared__ _Float16 lB[TP * DD];
    __shared__ float Cm[TP * TP];
    __shared__ float DPm[(TT + 1) * (TT + 1)];
    __shared__ float pw1[48], pb1[16], pw2[48], pb2[3], plg[3], plb[3];

    const int tid = threadIdx.x;
    const int b = blockIdx.x >> 8;
    const int r = blockIdx.x & 255;

    const uint32_t* srcA = (const uint32_t*)(qWh + (size_t)b * TP * DD);
    const uint32_t* srcB = (const uint32_t*)(krefh + (size_t)r * TP * DD);
    uint32_t* dA = (uint32_t*)lA;
    uint32_t* dB = (uint32_t*)lB;
    for (int i = tid; i < TP * DD / 2; i += 64) { dA[i] = srcA[i]; dB[i] = srcB[i]; }
    if (tid < 48) { pw1[tid] = nn2_w1[tid]; pw2[tid] = nn2_w2[tid]; }
    if (tid < 16) pb1[tid] = nn2_b1[tid];
    if (tid < 3)  { pb2[tid] = nn2_b2[tid]; plg[tid] = nn2_ln_g[tid]; plb[tid] = nn2_ln_b[tid]; }
    const float scale = nn1_scale[0];
    __syncthreads();

    // WMMA: C[t][u] = dot(qW[t,:32], kref[u,:32]); scaled by -nn1_scale.
    const int wave  = tid >> 5;
    const int lane  = tid & 31;
    const int laneM = lane & 15;
    const int hi    = lane >> 4;

    #pragma unroll
    for (int miI = 0; miI < 2; ++miI) {
        const int mi = wave * 2 + miI;
        // A fragment: 16x32 f16, lanes 0-15 hold K 0..15, lanes 16-31 hold K 16..31
        v16h a;
        const _Float16* Arow = lA + (mi * 16 + laneM) * DD;
        #pragma unroll
        for (int e = 0; e < 16; ++e) a[e] = Arow[hi * 16 + e];
        #pragma unroll
        for (int ni = 0; ni < 4; ++ni) {
            // B fragment: 32x16 f16, lane = column N; lanes 0-15 K=0..15, lanes 16-31 K=16..31
            v16h bf;
            const _Float16* Brow = lB + (ni * 16 + laneM) * DD;
            #pragma unroll
            for (int e = 0; e < 16; ++e) bf[e] = Brow[hi * 16 + e];
            v8f acc = {};
            acc = __builtin_amdgcn_wmma_f32_16x16x32_f16(
                false, a, false, bf, (short)0, acc, false, false);
            const int col = ni * 16 + laneM;
            #pragma unroll
            for (int i = 0; i < 8; ++i) {
                const int row = mi * 16 + i + hi * 8;
                Cm[row * TP + col] = -scale * acc[i];
            }
        }
    }
    __syncthreads();

    // DP init: DP[0][0]=0, first row/col = PAD_BIAS
    for (int i = tid; i < (TT + 1) * (TT + 1); i += 64) DPm[i] = 0.f;
    __syncthreads();
    if (tid < TT) { DPm[tid + 1] = 8.0f; DPm[(tid + 1) * (TT + 1)] = 8.0f; }
    __syncthreads();

    const int t = tid + 1;
    for (int kd = 2; kd <= 2 * TT; ++kd) {
        const int u = kd - t;
        if (t <= TT && u >= 1 && u <= TT) {
            const float n0 = DPm[(t - 1) * (TT + 1) + u];       // up
            const float n1 = DPm[t * (TT + 1) + u - 1];          // left
            const float n2 = DPm[(t - 1) * (TT + 1) + u - 1];    // diag
            // _nn2 with tau == 2 (min(max(exp,32),2) is identically 2)
            const float a0 = -0.5f * n0, a1 = -0.5f * n1, a2 = -0.5f * n2;
            const float mm = fmaxf(a0, fmaxf(a1, a2));
            const float lse = mm + logf(expf(a0 - mm) + expf(a1 - mm) + expf(a2 - mm));
            const float best = -2.f * lse;
            const float z0 = -fabsf(n0 - best), z1 = -fabsf(n1 - best), z2 = -fabsf(n2 - best);
            const float mu = (z0 + z1 + z2) * (1.f / 3.f);
            const float d0 = z0 - mu, d1 = z1 - mu, d2 = z2 - mu;
            const float var = (d0 * d0 + d1 * d1 + d2 * d2) * (1.f / 3.f);
            const float iv = rsqrtf(var + 1e-5f);
            const float x0 = d0 * iv * plg[0] + plb[0];
            const float x1 = d1 * iv * plg[1] + plb[1];
            const float x2 = d2 * iv * plg[2] + plb[2];
            float g0 = pb2[0], g1 = pb2[1], g2 = pb2[2];
            #pragma unroll
            for (int j = 0; j < 16; ++j) {
                float hh = pw1[j * 3 + 0] * x0 + pw1[j * 3 + 1] * x1 + pw1[j * 3 + 2] * x2 + pb1[j];
                hh = fmaxf(hh, 0.f);
                g0 += pw2[j] * hh; g1 += pw2[16 + j] * hh; g2 += pw2[32 + j] * hh;
            }
            const float gm = fmaxf(g0, fmaxf(g1, g2));
            const float e0 = expf(g0 - gm), e1 = expf(g1 - gm), e2 = expf(g2 - gm);
            const float isum = 1.f / (e0 + e1 + e2);
            DPm[t * (TT + 1) + u] =
                Cm[(t - 1) * TP + (u - 1)] + (n0 * e0 + n1 * e1 + n2 * e2) * isum;
        }
        __syncthreads();
    }
    if (tid == 0) flat[blockIdx.x] = DPm[TT * (TT + 1) + TT];
}

// ---------------------------------------------------------------------------
// Kernel 3: soft-sort (isotonic regression via LDS atomic max) + kNN readout.
// One block per batch row.
// ---------------------------------------------------------------------------
__device__ __forceinline__ unsigned ordf(float f) {
    int i = __float_as_int(f);
    return (i < 0) ? ~(unsigned)i : ((unsigned)i | 0x80000000u);
}
__device__ __forceinline__ float deordf(unsigned u) {
    int i = (u & 0x80000000u) ? (int)(u & 0x7fffffffu) : (int)(~u);
    return __int_as_float(i);
}

__global__ __launch_bounds__(256) void ainn_post_kernel(
    const float* __restrict__ flat,
    const float* __restrict__ phi_w1, const float* __restrict__ phi_b1,
    const float* __restrict__ phi_w2, const float* __restrict__ phi_b2,
    const float* __restrict__ rho_w1, const float* __restrict__ rho_b1,
    const float* __restrict__ rho_w2, const float* __restrict__ rho_b2,
    const float* __restrict__ res_scale, const float* __restrict__ res_scale2,
    float* __restrict__ out)
{
    __shared__ float    fl[RR];
    __shared__ float    asc[RR];
    __shared__ float    csum[RR + 1];
    __shared__ unsigned incb[RR];
    __shared__ float    yk[8];
    __shared__ float    ohm[64];
    __shared__ float    hph[8 * 32];

    const int b = blockIdx.x, tid = threadIdx.x;
    fl[tid]   = flat[b * RR + tid];
    incb[tid] = 0u;   // lowest ordered key
    __syncthreads();

    // rank-sort ascending (ties broken by index)
    {
        const float ai = fl[tid];
        int cnt = 0;
        for (int j = 0; j < RR; ++j) {
            const float aj = fl[j];
            cnt += (aj < ai) || (aj == ai && j < tid);
        }
        asc[cnt] = ai;
    }
    __syncthreads();

    // y = w + asc (w[i]=(256-i)/eps, eps=0.01); neg = -y; prefix sums
    if (tid == 0) {
        float c = 0.f; csum[0] = 0.f;
        for (int i = 0; i < RR; ++i) {
            const float w = (float)(RR - i) * 100.f;
            c += -(w + asc[i]);
            csum[i + 1] = c;
        }
    }
    __syncthreads();

    // isotonic (decreasing): inc[j] = max_{k<=j} min_{l>=j} mean(neg[k..l])
    {
        const int k = tid;
        const float ck = csum[k];
        float runmin = 3.0e38f;
        for (int j = RR - 1; j >= k; --j) {
            const float mean = (csum[j + 1] - ck) / (float)(j - k + 1);
            runmin = fminf(runmin, mean);
            atomicMax(&incb[j], ordf(runmin));
        }
    }
    __syncthreads();

    // sorted[j] = dual - w = -inc[j] - w[j]; yk = last KNN_K
    if (tid < 8) {
        const int j = RR - 8 + tid;
        yk[tid] = -deordf(incb[j]) - (float)(RR - j) * 100.f;
    }
    __syncthreads();

    // per-k soft label + one-hot + hphi MLP
    if (tid < 8) {
        const int ki = tid;
        const float ykv = yk[ki];
        float mx = -3.0e38f;
        for (int rr2 = 0; rr2 < RR; ++rr2)
            mx = fmaxf(mx, -fabsf(fl[rr2] - ykv) * 10.f);
        float se = 0.f, sl = 0.f;
        for (int rr2 = 0; rr2 < RR; ++rr2) {
            const float e = expf(-fabsf(fl[rr2] - ykv) * 10.f - mx);
            se += e; sl += e * (float)(rr2 >> 5);
        }
        const float lab = sl / se;
        float gv[8]; float gm = -3.0e38f;
        #pragma unroll
        for (int c = 0; c < 8; ++c) { gv[c] = -fabsf(lab - (float)c) * 1e5f; gm = fmaxf(gm, gv[c]); }
        float ssum = 0.f;
        #pragma unroll
        for (int c = 0; c < 8; ++c) { gv[c] = expf(gv[c] - gm); ssum += gv[c]; }
        #pragma unroll
        for (int c = 0; c < 8; ++c) ohm[ki * 8 + c] = gv[c] / ssum;
        float h1[32];
        #pragma unroll
        for (int h = 0; h < 32; ++h) h1[h] = gelu_f(ykv * phi_w1[h] + phi_b1[h]);
        for (int h = 0; h < 32; ++h) {
            float a2 = phi_b2[h];
            #pragma unroll
            for (int k2 = 0; k2 < 32; ++k2) a2 += h1[k2] * phi_w2[h * 32 + k2];
            hph[ki * 32 + h] = gelu_f(a2);
        }
    }
    __syncthreads();

    // per-class pooling + rho head
    if (tid < 8) {
        const int c = tid;
        float cs = 0.f;
        float pooled[33];
        #pragma unroll
        for (int h = 0; h < 33; ++h) pooled[h] = 0.f;
        for (int ki = 0; ki < 8; ++ki) {
            const float o = ohm[ki * 8 + c];
            cs += o * yk[ki];
            #pragma unroll
            for (int h = 0; h < 32; ++h) pooled[h] += o * hph[ki * 32 + h];
            pooled[32] += o;
        }
        float delta = rho_b2[0];
        for (int hh = 0; hh < 32; ++hh) {
            float a = rho_b1[hh];
            #pragma unroll
            for (int h = 0; h < 33; ++h) a += pooled[h] * rho_w1[hh * 33 + h];
            delta += gelu_f(a) * rho_w2[hh];
        }
        out[b * 8 + c] = cs * res_scale2[0] + res_scale[0] * delta;
    }
}

// ---------------------------------------------------------------------------
extern "C" void kernel_launch(void* const* d_in, const int* in_sizes, int n_in,
                              void* d_out, int out_size, void* d_ws, size_t ws_size,
                              hipStream_t stream) {
    (void)in_sizes; (void)n_in; (void)out_size; (void)ws_size;

    const float* x    = (const float*)d_in[0];
    const float* refs = (const float*)d_in[1];
    const float* prj_w = (const float*)d_in[2];
    const float* prj_b = (const float*)d_in[3];
    const float* b1_dw_w = (const float*)d_in[4];
    const float* b1_dw_b = (const float*)d_in[5];
    const float* b1_pw_w = (const float*)d_in[6];
    const float* b1_pw_b = (const float*)d_in[7];
    const float* b2_dw_w = (const float*)d_in[8];
    const float* b2_dw_b = (const float*)d_in[9];
    const float* b2_pw_w = (const float*)d_in[10];
    const float* b2_pw_b = (const float*)d_in[11];
    const float* b3_dw_w = (const float*)d_in[12];
    const float* b3_dw_b = (const float*)d_in[13];
    const float* b3_pw_w = (const float*)d_in[14];
    const float* b3_pw_b = (const float*)d_in[15];
    const float* ln_g = (const float*)d_in[16];
    const float* ln_b = (const float*)d_in[17];
    const float* nn1_W = (const float*)d_in[18];
    const float* nn1_scale = (const float*)d_in[19];
    /* d_in[20] nn2_log_tau: tau is identically 2.0 -> unused */
    const float* nn2_ln_g = (const float*)d_in[21];
    const float* nn2_ln_b = (const float*)d_in[22];
    const float* nn2_w1 = (const float*)d_in[23];
    const float* nn2_b1 = (const float*)d_in[24];
    const float* nn2_w2 = (const float*)d_in[25];
    const float* nn2_b2 = (const float*)d_in[26];
    const float* phi_w1 = (const float*)d_in[27];
    const float* phi_b1 = (const float*)d_in[28];
    const float* phi_w2 = (const float*)d_in[29];
    const float* phi_b2 = (const float*)d_in[30];
    const float* rho_w1 = (const float*)d_in[31];
    const float* rho_b1 = (const float*)d_in[32];
    const float* rho_w2 = (const float*)d_in[33];
    const float* rho_b2 = (const float*)d_in[34];
    const float* res_scale  = (const float*)d_in[35];
    const float* res_scale2 = (const float*)d_in[36];

    // workspace carve: qWh (8*64*32 f16) | krefh (256*64*32 f16) | flat (2048 f32)
    _Float16* qWh   = (_Float16*)d_ws;
    _Float16* krefh = qWh + (size_t)BSZ * TP * DD;
    float*    flat  = (float*)(krefh + (size_t)RR * TP * DD);
    float*    out   = (float*)d_out;

    ainn_enc_kernel<<<BSZ + RR, 128, 0, stream>>>(
        x, refs, prj_w, prj_b,
        b1_dw_w, b1_dw_b, b1_pw_w, b1_pw_b,
        b2_dw_w, b2_dw_b, b2_pw_w, b2_pw_b,
        b3_dw_w, b3_dw_b, b3_pw_w, b3_pw_b,
        ln_g, ln_b, nn1_W, qWh, krefh);

    ainn_dp_kernel<<<BSZ * RR, 64, 0, stream>>>(
        qWh, krefh, nn1_scale,
        nn2_ln_g, nn2_ln_b, nn2_w1, nn2_b1, nn2_w2, nn2_b2, flat);

    ainn_post_kernel<<<BSZ, 256, 0, stream>>>(
        flat, phi_w1, phi_b1, phi_w2, phi_b2,
        rho_w1, rho_b1, rho_w2, rho_b2, res_scale, res_scale2, out);
}
